// GATForDialog_18923625906415
// MI455X (gfx1250) — matrix operand order repaired and checked by reading, not compiled, not run on hardware
//
#include <hip/hip_runtime.h>

// ---------------------------------------------------------------------------
// GAT forward for MI455X (gfx1250): bf16 WMMA GEMMs + graph scatter kernels.
// ---------------------------------------------------------------------------

typedef __bf16 bf16_t;
typedef bf16_t v16bf __attribute__((ext_vector_type(16)));
typedef float  v8f   __attribute__((ext_vector_type(8)));

static constexpr int N_NODES = 20000;
static constexpr int E_EDGES = 320000;
static constexpr int ETOT    = E_EDGES + N_NODES;   // with self loops
static constexpr int F_INPUT = 1024;

// ------------------------------ helpers ------------------------------------

__device__ __forceinline__ unsigned short f2bf(float f) {
    unsigned int u = __float_as_uint(f);
    u += 0x7FFFu + ((u >> 16) & 1u);          // round-to-nearest-even
    return (unsigned short)(u >> 16);
}

__device__ __forceinline__ float atomicMaxF(float* addr, float val) {
    if (val >= 0.f)
        return __int_as_float(atomicMax((int*)addr, __float_as_int(val)));
    else
        return __uint_as_float(atomicMin((unsigned int*)addr, __float_as_uint(val)));
}

// ------------------------------ conversion ---------------------------------

__global__ void k_cvt_bf16(const float* __restrict__ src,
                           unsigned short* __restrict__ dst, int n) {
    int i = blockIdx.x * blockDim.x + threadIdx.x;
    if (i < n) dst[i] = f2bf(src[i]);
}

// W [K, Nn] f32 -> WT [Nn, K] bf16
__global__ void k_transpose_bf16(const float* __restrict__ w,
                                 unsigned short* __restrict__ wT, int K, int Nn) {
    int i = blockIdx.x * blockDim.x + threadIdx.x;
    if (i >= K * Nn) return;
    int k = i / Nn, n = i - k * Nn;
    wT[(size_t)n * K + k] = f2bf(w[i]);
}

__global__ void k_fill(float* __restrict__ p, float v, int n) {
    int i = blockIdx.x * blockDim.x + threadIdx.x;
    if (i < n) p[i] = v;
}

// ------------------------------ WMMA GEMM ----------------------------------
// C[M,Nn] = A[M,K](bf16) * BT[Nn,K]^T(bf16), f32 accumulate.
// Block: 256 threads = 8 waves; tile 128x64; 64 K-values per LDS stage,
// double-buffered => one barrier per 8 v_wmma_f32_16x16x32_bf16 per wave.

static constexpr int BM = 128;
static constexpr int BN = 64;
static constexpr int KSTAGE = 64;     // halfs of K per stage (2 WMMA k-steps)
static constexpr int A_STRIDE = 72;   // 64 + 8 pad halfs (bank spread)
static constexpr int B_STRIDE = 72;

__global__ __launch_bounds__(256)
void k_gemm_bf16(const unsigned short* __restrict__ A,
                 const unsigned short* __restrict__ BT,
                 float* __restrict__ C,
                 int M, int K, int Nn)
{
    __shared__ unsigned short lA[2][BM * A_STRIDE];   // 2 x 18432 B
    __shared__ unsigned short lB[2][BN * B_STRIDE];   // 2 x  9216 B

    const int tid  = threadIdx.x;
    const int lane = tid & 31;
    const int wave = tid >> 5;
    const int wm   = wave & 3;        // 0..3 along M (4*32 = 128)
    const int wn   = wave >> 2;       // 0..1 along N (2*32 = 64)
    const int lrow = lane & 15;
    const int hi   = lane >> 4;       // lane group (K-split per ISA layout)
    const int blockM = blockIdx.x * BM;
    const int blockN = blockIdx.y * BN;

    // global staging assignments
    const int arow = tid >> 1, achunk = tid & 1;   // A: 32 halfs / thread
    const int agrow = blockM + arow;
    const bool aval = agrow < M;
    const int bn = tid >> 2, bchunk = tid & 3;     // B: 16 halfs / thread
    const int bgn = blockN + bn;                   // Nn multiple of 64

    const uint4 z4 = make_uint4(0u, 0u, 0u, 0u);
    uint4 pa[4], pb[2];

    auto gload = [&](int k0) {
        if (aval) {
            const uint4* gp = (const uint4*)(A + (size_t)agrow * K + k0 + achunk * 32);
            pa[0] = gp[0]; pa[1] = gp[1]; pa[2] = gp[2]; pa[3] = gp[3];
        } else {
            pa[0] = z4; pa[1] = z4; pa[2] = z4; pa[3] = z4;
        }
        const uint4* gq = (const uint4*)(BT + (size_t)bgn * K + k0 + bchunk * 16);
        pb[0] = gq[0]; pb[1] = gq[1];
    };
    auto lstore = [&](int buf) {
        uint4* lp = (uint4*)(&lA[buf][arow * A_STRIDE + achunk * 32]);
        lp[0] = pa[0]; lp[1] = pa[1]; lp[2] = pa[2]; lp[3] = pa[3];
        uint4* lq = (uint4*)(&lB[buf][bn * B_STRIDE + bchunk * 16]);
        lq[0] = pb[0]; lq[1] = pb[1];
    };

    v8f acc[2][2] = {};

    gload(0);
    lstore(0);
    __syncthreads();

    int cur = 0;
    for (int k0 = 0; k0 < K; k0 += KSTAGE) {
        const bool nxt = (k0 + KSTAGE) < K;
        if (nxt) gload(k0 + KSTAGE);   // prefetch next stage into registers

        #pragma unroll
        for (int s = 0; s < 2; ++s) {
            // A fragments: 16x32 bf16, slots 0-7 -> k = hi*8+i, 8-15 -> +16
            v16bf af[2];
            #pragma unroll
            for (int tm = 0; tm < 2; ++tm) {
                const unsigned short* base =
                    &lA[cur][(wm * 32 + tm * 16 + lrow) * A_STRIDE + s * 32 + hi * 8];
                union { uint4 u[2]; v16bf v; } cv;
                cv.u[0] = *(const uint4*)(base);
                cv.u[1] = *(const uint4*)(base + 16);
                af[tm] = cv.v;
            }
            // B fragments: 32x16 bf16, slots 0-7 -> k = hi*8+i, 8-15 -> +16
            v16bf bfg[2];
            #pragma unroll
            for (int tn = 0; tn < 2; ++tn) {
                const unsigned short* base =
                    &lB[cur][(wn * 32 + tn * 16 + lrow) * B_STRIDE + s * 32 + hi * 8];
                union { uint4 u[2]; v16bf v; } cv;
                cv.u[0] = *(const uint4*)(base);
                cv.u[1] = *(const uint4*)(base + 16);
                bfg[tn] = cv.v;
            }
            #pragma unroll
            for (int tm = 0; tm < 2; ++tm)
                #pragma unroll
                for (int tn = 0; tn < 2; ++tn)
                    acc[tm][tn] = __builtin_amdgcn_wmma_f32_16x16x32_bf16(
                        false, af[tm], false, bfg[tn],
                        (short)0, acc[tm][tn], false, false);
        }

        if (nxt) {
            lstore(cur ^ 1);           // fill the other buffer
            __syncthreads();
            cur ^= 1;
        }
    }

    // ---- store: C/D layout — VGPR r, lanes 0-15 -> M=r, lanes 16-31 -> M=8+r
    if (blockM + BM <= M) {            // hot path: no bounds checks
        #pragma unroll
        for (int tm = 0; tm < 2; ++tm) {
            const int row0 = blockM + wm * 32 + tm * 16 + hi * 8;
            #pragma unroll
            for (int tn = 0; tn < 2; ++tn) {
                const int col = blockN + wn * 32 + tn * 16 + lrow;
                #pragma unroll
                for (int r = 0; r < 8; ++r)
                    C[(size_t)(row0 + r) * Nn + col] = acc[tm][tn][r];
            }
        }
    } else {
        #pragma unroll
        for (int tm = 0; tm < 2; ++tm) {
            const int row0 = blockM + wm * 32 + tm * 16 + hi * 8;
            #pragma unroll
            for (int tn = 0; tn < 2; ++tn) {
                const int col = blockN + wn * 32 + tn * 16 + lrow;
                #pragma unroll
                for (int r = 0; r < 8; ++r)
                    if (row0 + r < M)
                        C[(size_t)(row0 + r) * Nn + col] = acc[tm][tn][r];
            }
        }
    }
}

// ------------------------------ attention ----------------------------------

// al_s[n,h] = <h[n,h,:], a_src[h,:]> ; al_d likewise.
__global__ void k_compute_al(const float* __restrict__ h,
                             const float* __restrict__ a_s,
                             const float* __restrict__ a_d,
                             float* __restrict__ als, float* __restrict__ ald,
                             int n, int Hh, int Cc) {
    int i = blockIdx.x * blockDim.x + threadIdx.x;
    if (i >= n * Hh) return;
    int node = i / Hh, hd = i - node * Hh;
    const float* hp = h + ((size_t)node * Hh + hd) * Cc;
    const float* as = a_s + hd * Cc;
    const float* ad = a_d + hd * Cc;
    float s = 0.f, d = 0.f;
    for (int c = 0; c < Cc; ++c) { float v = hp[c]; s += v * as[c]; d += v * ad[c]; }
    als[i] = s;
    ald[i] = d;
}

// e = leaky_relu(al_s[src]+al_d[dst]); segment max over dst.
__global__ void k_edge_max(const int* __restrict__ ei,
                           const float* __restrict__ als, const float* __restrict__ ald,
                           float* __restrict__ ebuf, float* __restrict__ emax, int Hh) {
    int idx = blockIdx.x * blockDim.x + threadIdx.x;
    if (idx >= ETOT * Hh) return;
    int e = idx / Hh, hd = idx - e * Hh;
    int src, dst;
    if (e < E_EDGES) { src = ei[e]; dst = ei[E_EDGES + e]; }
    else             { src = dst = e - E_EDGES; }
    float v = als[src * Hh + hd] + ald[dst * Hh + hd];
    v = (v > 0.f) ? v : v * 0.2f;                 // NEG_SLOPE
    ebuf[idx] = v;
    atomicMaxF(&emax[dst * Hh + hd], v);
}

// ex = exp(e - emax[dst]); segment sum over dst.
__global__ void k_edge_exp(const int* __restrict__ ei,
                           float* __restrict__ ebuf, const float* __restrict__ emax,
                           float* __restrict__ denom, int Hh) {
    int idx = blockIdx.x * blockDim.x + threadIdx.x;
    if (idx >= ETOT * Hh) return;
    int e = idx / Hh, hd = idx - e * Hh;
    int dst = (e < E_EDGES) ? ei[E_EDGES + e] : (e - E_EDGES);
    float ex = __expf(ebuf[idx] - emax[dst * Hh + hd]);
    ebuf[idx] = ex;
    atomicAdd(&denom[dst * Hh + hd], ex);
}

// agg[dst,h,:] += h[src,h,:] * alpha ; one thread per (edge, head, float4).
__global__ void k_edge_agg(const int* __restrict__ ei,
                           const float* __restrict__ hbuf,
                           const float* __restrict__ ebuf, const float* __restrict__ denom,
                           float* __restrict__ agg, int Hh, int Cc) {
    int idx = blockIdx.x * blockDim.x + threadIdx.x;
    int cq = Cc >> 2;
    if (idx >= ETOT * Hh * cq) return;
    int c4 = idx % cq;
    int t  = idx / cq;
    int hd = t % Hh;
    int e  = t / Hh;
    int src, dst;
    if (e < E_EDGES) { src = ei[e]; dst = ei[E_EDGES + e]; }
    else             { src = dst = e - E_EDGES; }
    float alpha = ebuf[e * Hh + hd] / (denom[dst * Hh + hd] + 1e-16f);
    const float4 m = *(const float4*)(hbuf + ((size_t)src * Hh + hd) * Cc + c4 * 4);
    float* op = agg + ((size_t)dst * Hh + hd) * Cc + c4 * 4;
    atomicAdd(op + 0, m.x * alpha);
    atomicAdd(op + 1, m.y * alpha);
    atomicAdd(op + 2, m.z * alpha);
    atomicAdd(op + 3, m.w * alpha);
}

// out = (relu?)(agg + bias); also emit bf16 copy for next layer's GEMM input.
__global__ void k_finalize(float* __restrict__ agg, const float* __restrict__ bias,
                           unsigned short* __restrict__ outbf, int HC, int relu) {
    int i = blockIdx.x * blockDim.x + threadIdx.x;
    if (i >= N_NODES * HC) return;
    float v = agg[i] + bias[i % HC];
    if (relu) v = fmaxf(v, 0.f);
    agg[i] = v;
    if (outbf) outbf[i] = f2bf(v);
}

// out[n,j] = <h3[n,:], wc[:,j]> + bc[j]  (C=64, NC=4)
__global__ void k_classifier(const float* __restrict__ h,
                             const float* __restrict__ wc,
                             const float* __restrict__ bc,
                             float* __restrict__ out) {
    int i = blockIdx.x * blockDim.x + threadIdx.x;
    if (i >= N_NODES * 4) return;
    int n = i >> 2, j = i & 3;
    const float* hp = h + (size_t)n * 64;
    float s = bc[j];
    for (int c = 0; c < 64; ++c) s += hp[c] * wc[c * 4 + j];
    out[i] = s;
}

// ------------------------------ launcher -----------------------------------

extern "C" void kernel_launch(void* const* d_in, const int* in_sizes, int n_in,
                              void* d_out, int out_size, void* d_ws, size_t ws_size,
                              hipStream_t stream) {
    const float* x   = (const float*)d_in[0];
    const int*   ei  = (const int*)d_in[1];
    const float* w1  = (const float*)d_in[2];
    const float* as1 = (const float*)d_in[3];
    const float* ad1 = (const float*)d_in[4];
    const float* b1  = (const float*)d_in[5];
    const float* w2  = (const float*)d_in[6];
    const float* as2 = (const float*)d_in[7];
    const float* ad2 = (const float*)d_in[8];
    const float* b2  = (const float*)d_in[9];
    const float* w3  = (const float*)d_in[10];
    const float* as3 = (const float*)d_in[11];
    const float* ad3 = (const float*)d_in[12];
    const float* b3  = (const float*)d_in[13];
    const float* wc  = (const float*)d_in[14];
    const float* bc  = (const float*)d_in[15];
    float* out = (float*)d_out;

    // workspace carve-up (all chunk sizes multiples of 256 B)
    char* p = (char*)d_ws;
    unsigned short* inbf = (unsigned short*)p; p += (size_t)N_NODES * 1024 * 2;  // GEMM input bf16
    unsigned short* wT   = (unsigned short*)p; p += (size_t)256 * 1024 * 2;      // transposed W bf16
    float* hbuf  = (float*)p; p += (size_t)N_NODES * 256 * 4;                    // x@W
    float* als   = (float*)p; p += (size_t)N_NODES * 4 * 4;
    float* ald   = (float*)p; p += (size_t)N_NODES * 4 * 4;
    float* emax  = (float*)p; p += (size_t)N_NODES * 4 * 4;
    float* denom = (float*)p; p += (size_t)N_NODES * 4 * 4;
    float* ebuf  = (float*)p; p += (size_t)ETOT * 4 * 4;
    float* agg   = (float*)p; p += (size_t)N_NODES * 256 * 4;                    // layer output

    auto run_layer = [&](const unsigned short* Abf, int Kin, const float* W,
                         const float* a_s, const float* a_d, const float* bias,
                         int Hh, int Cc, bool relu, unsigned short* nextbf) {
        int HC = Hh * Cc;
        int wtot = Kin * HC;
        k_transpose_bf16<<<(wtot + 255) / 256, 256, 0, stream>>>(W, wT, Kin, HC);
        dim3 g((N_NODES + BM - 1) / BM, HC / BN);
        k_gemm_bf16<<<g, 256, 0, stream>>>(Abf, wT, hbuf, N_NODES, Kin, HC);
        k_compute_al<<<(N_NODES * Hh + 255) / 256, 256, 0, stream>>>(
            hbuf, a_s, a_d, als, ald, N_NODES, Hh, Cc);
        k_fill<<<(N_NODES * HC + 255) / 256, 256, 0, stream>>>(agg, 0.f, N_NODES * HC);
        k_fill<<<(N_NODES * Hh + 255) / 256, 256, 0, stream>>>(denom, 0.f, N_NODES * Hh);
        k_fill<<<(N_NODES * Hh + 255) / 256, 256, 0, stream>>>(emax, -3.402823466e38f, N_NODES * Hh);
        int nEH = ETOT * Hh;
        k_edge_max<<<(nEH + 255) / 256, 256, 0, stream>>>(ei, als, ald, ebuf, emax, Hh);
        k_edge_exp<<<(nEH + 255) / 256, 256, 0, stream>>>(ei, ebuf, emax, denom, Hh);
        int nAgg = ETOT * Hh * (Cc / 4);
        k_edge_agg<<<(nAgg + 255) / 256, 256, 0, stream>>>(ei, hbuf, ebuf, denom, agg, Hh, Cc);
        k_finalize<<<(N_NODES * HC + 255) / 256, 256, 0, stream>>>(
            agg, bias, nextbf, HC, relu ? 1 : 0);
    };

    // layer 1: F_IN -> 4*64, relu
    k_cvt_bf16<<<(N_NODES * F_INPUT + 255) / 256, 256, 0, stream>>>(x, inbf, N_NODES * F_INPUT);
    run_layer(inbf, F_INPUT, w1, as1, ad1, b1, 4, 64, true, inbf);
    // layer 2: 256 -> 4*64, relu
    run_layer(inbf, 256, w2, as2, ad2, b2, 4, 64, true, inbf);
    // layer 3: 256 -> 64 (1 head, mean == identity), no relu
    run_layer(inbf, 256, w3, as3, ad3, b3, 1, 64, false, nullptr);
    // classifier: [N,64] @ [64,4] + bc
    k_classifier<<<(N_NODES * 4 + 255) / 256, 256, 0, stream>>>(agg, wc, bc, out);
}
